// CLIPTextBlock_5892695130388
// MI455X (gfx1250) — matrix operand order repaired and checked by reading, not compiled, not run on hardware
//
#include <hip/hip_runtime.h>
#include <hip/hip_bf16.h>

#define BB 2
#define TT 2048
#define DD 768
#define HH 12
#define DH 64
#define FF_ 3072
#define MM (BB*TT)          // 4096 rows
#define NEGBIG -1.0e30f

typedef __attribute__((ext_vector_type(16))) __bf16 v16bf;
typedef __attribute__((ext_vector_type(8)))  float  v8f;
typedef __attribute__((ext_vector_type(4)))  unsigned int su4;
typedef __attribute__((ext_vector_type(8)))  unsigned int su8;

union FragBF { v16bf v; unsigned int u[8]; unsigned short s[16]; };

static __device__ __forceinline__ unsigned short f32_to_bf16(float f) {
  unsigned int u = __float_as_uint(f);
  unsigned int r = 0x7FFFu + ((u >> 16) & 1u);
  return (unsigned short)((u + r) >> 16);
}

static __device__ __forceinline__ v8f vzero8() {
  v8f z = {0.f,0.f,0.f,0.f,0.f,0.f,0.f,0.f};
  return z;
}

static __device__ __forceinline__ float gelu_t(float x) {
  float x3 = x * x * x;
  return 0.5f * x * (1.0f + tanhf(0.7978845608028654f * (x + 0.044715f * x3)));
}

// A-fragment (16x32 bf16, M x K): lane m=lane&15 is the row; VGPR r holds K pair
// k = (r<4 ? 2r : 16+2(r-4)) + 8*(lane>>4)   [ISA 7.12.2, 16-bit A 16x32]
static __device__ __forceinline__ v16bf load_a_frag(const unsigned short* base, int stride, int lane) {
  const int m = lane & 15, kh = lane >> 4;
  FragBF f;
  const unsigned short* rp = base + m * stride;
#pragma unroll
  for (int r = 0; r < 8; ++r) {
    int k = ((r < 4) ? (2 * r) : (16 + 2 * (r - 4))) + 8 * kh;
    f.u[r] = *(const unsigned int*)(rp + k);
  }
  return f.v;
}

// B-fragment (32x16 bf16, K x N) loaded from a TRANSPOSED LDS image B_s[n][k]:
// lane n=lane&15 is the column; VGPR r holds K pair k = (r<4 ? 2r : 16+2(r-4)) + 8*(lane>>4)
static __device__ __forceinline__ v16bf load_bT_frag(const unsigned short* base, int stride, int lane) {
  const int n = lane & 15, kh = lane >> 4;
  FragBF f;
  const unsigned short* rp = base + n * stride;
#pragma unroll
  for (int r = 0; r < 8; ++r) {
    int k = ((r < 4) ? (2 * r) : (16 + 2 * (r - 4))) + 8 * kh;
    f.u[r] = *(const unsigned int*)(rp + k);
  }
  return f.v;
}

// ---- Tensor Data Mover: 2D tile load Global->LDS (D# per ISA 08, data_size=2B) ----
// pad_interval code 4 = 32 dwords (one 64-elem bf16 row), pad_amount code 7 = 8 dwords
// -> LDS row stride of 80 ushorts, matching the padded LDS tiles below.
static __device__ __forceinline__ void tdm_load_2d_bf16(
    unsigned lds_addr, unsigned long long gaddr,
    unsigned tensor_d0, unsigned tensor_d1, unsigned tile_d0, unsigned tile_d1,
    unsigned stride_d0, unsigned pad_int_code, unsigned pad_amt_code, unsigned pad_en) {
  su4 g0;
  g0[0] = 1u;                                          // count=1 (valid), user mode
  g0[1] = lds_addr;                                    // lds_addr (bytes)
  g0[2] = (unsigned)(gaddr & 0xFFFFFFFFu);             // global_addr[31:0]
  g0[3] = (unsigned)((gaddr >> 32) & 0x01FFFFFFu) | (2u << 30);  // addr[56:32], type=2
  su8 g1;
  g1[0] = (1u << 16) | (pad_en << 20) | (pad_int_code << 22) | (pad_amt_code << 25);
  g1[1] = (tensor_d0 & 0xFFFFu) << 16;                 // tensor_dim0[15:0] @ bits 63:48
  g1[2] = ((tensor_d0 >> 16) & 0xFFFFu) | ((tensor_d1 & 0xFFFFu) << 16);
  g1[3] = ((tensor_d1 >> 16) & 0xFFFFu) | ((tile_d0 & 0xFFFFu) << 16); // tile_dim0
  g1[4] = tile_d1 & 0xFFFFu;                           // tile_dim1 (tile_dim2=0)
  g1[5] = stride_d0;                                   // tensor_dim0_stride[31:0]
  g1[6] = 0u;
  g1[7] = 0u;
  su4 gz = {0u, 0u, 0u, 0u};
  asm volatile("tensor_load_to_lds %0, %1, %2, %3"
               :: "s"(g0), "s"(g1), "s"(gz), "s"(gz)
               : "memory");
}

// ---------------- f32 -> bf16 conversion ----------------
__global__ void cvt_bf16_kernel(const float* __restrict__ in, unsigned short* __restrict__ out, int n) {
  int i = blockIdx.x * blockDim.x + threadIdx.x;
  if (i < n) out[i] = f32_to_bf16(in[i]);
}

// ---------------- LayerNorm (row of 768), outputs bf16 ----------------
__global__ __launch_bounds__(256) void ln_kernel(const float* __restrict__ x,
                                                 const float* __restrict__ g,
                                                 const float* __restrict__ beta,
                                                 unsigned short* __restrict__ out) {
  const int row = blockIdx.x;
  const int tid = threadIdx.x, lane = tid & 31, wid = tid >> 5;
  const float* xr = x + (size_t)row * DD;
  float v0 = xr[tid], v1 = xr[tid + 256], v2 = xr[tid + 512];
  float s = v0 + v1 + v2;
#pragma unroll
  for (int m = 16; m >= 1; m >>= 1) s += __shfl_xor(s, m, 32);
  __shared__ float red[8], red2[8];
  if (lane == 0) red[wid] = s;
  __syncthreads();
  float tot = 0.f;
#pragma unroll
  for (int i = 0; i < 8; ++i) tot += red[i];
  float mu = tot * (1.0f / (float)DD);
  float d0 = v0 - mu, d1 = v1 - mu, d2 = v2 - mu;
  float s2 = d0 * d0 + d1 * d1 + d2 * d2;
#pragma unroll
  for (int m = 16; m >= 1; m >>= 1) s2 += __shfl_xor(s2, m, 32);
  if (lane == 0) red2[wid] = s2;
  __syncthreads();
  float tot2 = 0.f;
#pragma unroll
  for (int i = 0; i < 8; ++i) tot2 += red2[i];
  float rs = rsqrtf(tot2 * (1.0f / (float)DD) + 1e-5f);
  unsigned short* orow = out + (size_t)row * DD;
  orow[tid]       = f32_to_bf16(d0 * rs * g[tid]       + beta[tid]);
  orow[tid + 256] = f32_to_bf16(d1 * rs * g[tid + 256] + beta[tid + 256]);
  orow[tid + 512] = f32_to_bf16(d2 * rs * g[tid + 512] + beta[tid + 512]);
}

// ---------------- generic bf16 WMMA GEMM: C = A[MxK] * W[KxN] (+bias, epilogue) ----
// A tile staged by the Tensor Data Mover (wave 0 issues, TENSORcnt fence);
// B tile staged transposed by all threads (TDM cannot transpose).
// EPI 0: bias -> bf16 out   EPI 1: bias + gelu -> bf16 out   EPI 2: bias + resid -> f32 out
template <int EPI>
__global__ __launch_bounds__(256) void gemm_bf16_kernel(
    const unsigned short* __restrict__ A, const unsigned short* __restrict__ W,
    const float* __restrict__ bias, const float* __restrict__ resid,
    float* __restrict__ outF, unsigned short* __restrict__ outB,
    int M, int N, int K) {
  __shared__ unsigned short As[128][80];   // 64 data + 16 pad ushorts (TDM pad fields)
  __shared__ unsigned short Bs[128][74];   // TRANSPOSED tile [n][k]; 74 -> odd dword stride
  const int tid = threadIdx.x, lane = tid & 31, wid = tid >> 5;
  const int waveM = wid & 3, waveN = wid >> 2;
  const int m0 = blockIdx.y * 128, n0 = blockIdx.x * 128;
  const unsigned asBase = (unsigned)(size_t)&As[0][0];

  v8f acc[2][4];
#pragma unroll
  for (int i = 0; i < 2; ++i)
#pragma unroll
    for (int j = 0; j < 4; ++j) acc[i][j] = vzero8();

  for (int k0 = 0; k0 < K; k0 += 64) {
    __syncthreads();
    // A tile 128 rows x 64 bf16 via TDM (one DMA, wave 0 only)
    if (wid == 0) {
      tdm_load_2d_bf16(asBase,
                       (unsigned long long)(A + (size_t)m0 * K + k0),
                       (unsigned)K, (unsigned)M, 64u, 128u, (unsigned)K,
                       4u, 7u, 1u);
    }
    // fill B tile transposed: global rows k, cols n -> Bs[n][k]
#pragma unroll
    for (int t = 0; t < 4; ++t) {
      int s = tid + t * 256;
      int kr = s >> 4, c8 = (s & 15) * 8;
      uint4 v = *(const uint4*)(W + (size_t)(k0 + kr) * N + n0 + c8);
      const unsigned short* vs = (const unsigned short*)&v;
#pragma unroll
      for (int j = 0; j < 8; ++j) Bs[c8 + j][kr] = vs[j];
    }
    if (k0 + 64 < K) {  // hint next W K-block into caches (global_prefetch_b8)
      __builtin_prefetch(W + (size_t)(k0 + 64 + (tid & 63)) * N + n0, 0, 0);
    }
    if (wid == 0) __builtin_amdgcn_s_wait_tensorcnt(0);
    __syncthreads();
#pragma unroll
    for (int ks = 0; ks < 2; ++ks) {
      v16bf a0 = load_a_frag(&As[waveM * 32][ks * 32], 80, lane);
      v16bf a1 = load_a_frag(&As[waveM * 32 + 16][ks * 32], 80, lane);
      v16bf b[4];
#pragma unroll
      for (int j = 0; j < 4; ++j) b[j] = load_bT_frag(&Bs[waveN * 64 + j * 16][ks * 32], 74, lane);
#pragma unroll
      for (int j = 0; j < 4; ++j) {
        acc[0][j] = __builtin_amdgcn_wmma_f32_16x16x32_bf16(false, a0, false, b[j], (short)0, acc[0][j], false, false);
        acc[1][j] = __builtin_amdgcn_wmma_f32_16x16x32_bf16(false, a1, false, b[j], (short)0, acc[1][j], false, false);
      }
    }
  }
  // epilogue: C layout -> row = r + 8*(lane>>4), col = lane&15
  const int kh = lane >> 4, nl = lane & 15;
#pragma unroll
  for (int i = 0; i < 2; ++i)
#pragma unroll
    for (int j = 0; j < 4; ++j) {
      int col = n0 + waveN * 64 + j * 16 + nl;
      float bv = bias[col];
#pragma unroll
      for (int r = 0; r < 8; ++r) {
        int row = m0 + waveM * 32 + i * 16 + r + 8 * kh;
        float val = acc[i][j][r] + bv;
        if (EPI == 1) val = gelu_t(val);
        size_t idx = (size_t)row * N + col;
        if (EPI == 2) outF[idx] = val + resid[idx];
        else          outB[idx] = f32_to_bf16(val);
      }
    }
}

// ---------------- flash attention (causal), bf16 q/k/v [B,T,H,DH] ----------------
__global__ __launch_bounds__(128) void attn_kernel(
    const unsigned short* __restrict__ Q, const unsigned short* __restrict__ Kg,
    const unsigned short* __restrict__ Vg, unsigned short* __restrict__ O) {
  __shared__ unsigned short Ks[32][80];      // K tile [key][dh]  (== transposed-B image for scores)
  __shared__ unsigned short Vs[64][42];      // V tile transposed [dh][key]; odd dword stride
  __shared__ unsigned short Ps[4][16][40];   // per-wave P (16 q x 32 k) bf16
  const int tid = threadIdx.x, lane = tid & 31, wid = tid >> 5;
  const int bh = blockIdx.y, b = bh / HH, h = bh % HH;
  const int qb = blockIdx.x * 64;
  const int qrow0 = qb + wid * 16;
  const int m = lane & 15, kh = lane >> 4, nl = lane & 15;
  const unsigned short* Qb = Q  + ((size_t)b * TT) * DD + (size_t)h * DH;
  const unsigned short* Kb = Kg + ((size_t)b * TT) * DD + (size_t)h * DH;
  const unsigned short* Vb = Vg + ((size_t)b * TT) * DD + (size_t)h * DH;

  // Q fragments for this wave's 16 rows: A-layout 16x64 -> 2 frags (held in VGPRs)
  FragBF qf[2];
#pragma unroll
  for (int f = 0; f < 2; ++f) {
    const unsigned short* qp = Qb + (size_t)(qrow0 + m) * DD + f * 32;
#pragma unroll
    for (int r = 0; r < 8; ++r) {
      int k = ((r < 4) ? (2 * r) : (16 + 2 * (r - 4))) + 8 * kh;
      qf[f].u[r] = *(const unsigned int*)(qp + k);
    }
  }

  v8f o[4];
#pragma unroll
  for (int j = 0; j < 4; ++j) o[j] = vzero8();
  float mrow[8], lrow[8];
#pragma unroll
  for (int r = 0; r < 8; ++r) { mrow[r] = NEGBIG; lrow[r] = 0.f; }

  const int nkt = qb / 32 + 2;   // kv tiles of 32 keys, up through the diagonal
  for (int kt = 0; kt < nkt; ++kt) {
    const int k0 = kt * 32;
    __syncthreads();
    // stage K tile via async global->LDS (ASYNCcnt), V tile transposed manually
#pragma unroll
    for (int t = 0; t < 2; ++t) {
      int s = tid + t * 128;
      int key = s >> 3, c8 = (s & 7) * 8;
      unsigned laddr = (unsigned)(size_t)&Ks[key][c8];
      unsigned long long gaddr = (unsigned long long)(Kb + (size_t)(k0 + key) * DD + c8);
      asm volatile("global_load_async_to_lds_b128 %0, %1, off"
                   :: "v"(laddr), "v"(gaddr) : "memory");
      uint4 vv = *(const uint4*)(Vb + (size_t)(k0 + key) * DD + c8);
      const unsigned short* vs = (const unsigned short*)&vv;
#pragma unroll
      for (int j = 0; j < 8; ++j) Vs[c8 + j][key] = vs[j];
    }
    asm volatile("s_wait_asynccnt 0x0" ::: "memory");
    __syncthreads();

    // S = Q (16x64) . K^T -> two 16x16 tiles, contraction dh in 2 steps of 32
    v8f s0 = vzero8(), s1 = vzero8();
#pragma unroll
    for (int st = 0; st < 2; ++st) {
      v16bf bk0 = load_bT_frag(&Ks[0][st * 32], 80, lane);
      v16bf bk1 = load_bT_frag(&Ks[16][st * 32], 80, lane);
      s0 = __builtin_amdgcn_wmma_f32_16x16x32_bf16(false, qf[st].v, false, bk0, (short)0, s0, false, false);
      s1 = __builtin_amdgcn_wmma_f32_16x16x32_bf16(false, qf[st].v, false, bk1, (short)0, s1, false, false);
    }

    // online softmax per row (row r+8*kh lives entirely in one 16-lane half)
#pragma unroll
    for (int r = 0; r < 8; ++r) {
      int qrow = qrow0 + r + 8 * kh;
      float e0 = s0[r] * 0.125f; if (k0 + nl > qrow)      e0 = NEGBIG;
      float e1 = s1[r] * 0.125f; if (k0 + 16 + nl > qrow) e1 = NEGBIG;
      float mx = fmaxf(e0, e1);
#pragma unroll
      for (int msk = 8; msk >= 1; msk >>= 1) mx = fmaxf(mx, __shfl_xor(mx, msk, 32));
      float mnew = fmaxf(mrow[r], mx);
      float alpha = __expf(mrow[r] - mnew);
      float p0 = __expf(e0 - mnew), p1 = __expf(e1 - mnew);
      float ps = p0 + p1;
#pragma unroll
      for (int msk = 8; msk >= 1; msk >>= 1) ps += __shfl_xor(ps, msk, 32);
      lrow[r] = lrow[r] * alpha + ps;
      mrow[r] = mnew;
#pragma unroll
      for (int j = 0; j < 4; ++j) o[j][r] *= alpha;
      Ps[wid][r + 8 * kh][nl]      = f32_to_bf16(p0);
      Ps[wid][r + 8 * kh][16 + nl] = f32_to_bf16(p1);
    }

    // O += P (16x32) . V (32x64); P reloaded from per-wave LDS in A-layout
    FragBF pf;
    {
      const unsigned short* pp = &Ps[wid][0][0] + m * 40;
#pragma unroll
      for (int r = 0; r < 8; ++r) {
        int k = ((r < 4) ? (2 * r) : (16 + 2 * (r - 4))) + 8 * kh;
        pf.u[r] = *(const unsigned int*)(pp + k);
      }
    }
#pragma unroll
    for (int j = 0; j < 4; ++j) {
      v16bf bv = load_bT_frag(&Vs[j * 16][0], 42, lane);
      o[j] = __builtin_amdgcn_wmma_f32_16x16x32_bf16(false, pf.v, false, bv, (short)0, o[j], false, false);
    }
  }

  // normalize and store bf16 O[b,t,h,dh]
#pragma unroll
  for (int j = 0; j < 4; ++j) {
    int dh = j * 16 + nl;
#pragma unroll
    for (int r = 0; r < 8; ++r) {
      int qrow = qrow0 + r + 8 * kh;
      float val = o[j][r] / lrow[r];
      O[((size_t)(b * TT + qrow)) * DD + h * DH + dh] = f32_to_bf16(val);
    }
  }
}

// ---------------------------------------------------------------------------
extern "C" void kernel_launch(void* const* d_in, const int* in_sizes, int n_in,
                              void* d_out, int out_size, void* d_ws, size_t ws_size,
                              hipStream_t stream) {
  (void)in_sizes; (void)n_in; (void)out_size; (void)ws_size;
  const float* x     = (const float*)d_in[0];
  const float* ln1_g = (const float*)d_in[1];
  const float* ln1_b = (const float*)d_in[2];
  const float* wq    = (const float*)d_in[3];
  const float* bq    = (const float*)d_in[4];
  const float* wk    = (const float*)d_in[5];
  const float* bk    = (const float*)d_in[6];
  const float* wv    = (const float*)d_in[7];
  const float* bv    = (const float*)d_in[8];
  const float* wo    = (const float*)d_in[9];
  const float* bo    = (const float*)d_in[10];
  const float* ln2_g = (const float*)d_in[11];
  const float* ln2_b = (const float*)d_in[12];
  const float* w1    = (const float*)d_in[13];
  const float* b1    = (const float*)d_in[14];
  const float* w2    = (const float*)d_in[15];
  const float* b2    = (const float*)d_in[16];
  float* out = (float*)d_out;

  char* p = (char*)d_ws;
  auto carve = [&](size_t bytes) -> void* {
    void* r = (void*)p;
    p += (bytes + 255) & ~(size_t)255;
    return r;
  };
  unsigned short* wq_bf = (unsigned short*)carve((size_t)DD * DD * 2);
  unsigned short* wk_bf = (unsigned short*)carve((size_t)DD * DD * 2);
  unsigned short* wv_bf = (unsigned short*)carve((size_t)DD * DD * 2);
  unsigned short* wo_bf = (unsigned short*)carve((size_t)DD * DD * 2);
  unsigned short* w1_bf = (unsigned short*)carve((size_t)DD * FF_ * 2);
  unsigned short* w2_bf = (unsigned short*)carve((size_t)FF_ * DD * 2);
  unsigned short* h_bf  = (unsigned short*)carve((size_t)MM * DD * 2);
  unsigned short* q_bf  = (unsigned short*)carve((size_t)MM * DD * 2);
  unsigned short* k_bf  = (unsigned short*)carve((size_t)MM * DD * 2);
  unsigned short* v_bf  = (unsigned short*)carve((size_t)MM * DD * 2);
  unsigned short* o_bf  = (unsigned short*)carve((size_t)MM * DD * 2);
  float*          x1    = (float*)carve((size_t)MM * DD * 4);
  // q/k/v/o are contiguous, each MM*DD*2 and 256B-aligned -> reuse as MM x FF bf16
  unsigned short* m1_bf = q_bf;

  const int nDD = DD * DD, nDF = DD * FF_;
  cvt_bf16_kernel<<<(nDD + 255) / 256, 256, 0, stream>>>(wq, wq_bf, nDD);
  cvt_bf16_kernel<<<(nDD + 255) / 256, 256, 0, stream>>>(wk, wk_bf, nDD);
  cvt_bf16_kernel<<<(nDD + 255) / 256, 256, 0, stream>>>(wv, wv_bf, nDD);
  cvt_bf16_kernel<<<(nDD + 255) / 256, 256, 0, stream>>>(wo, wo_bf, nDD);
  cvt_bf16_kernel<<<(nDF + 255) / 256, 256, 0, stream>>>(w1, w1_bf, nDF);
  cvt_bf16_kernel<<<(nDF + 255) / 256, 256, 0, stream>>>(w2, w2_bf, nDF);

  ln_kernel<<<MM, 256, 0, stream>>>(x, ln1_g, ln1_b, h_bf);

  dim3 g768(DD / 128, MM / 128);    // (6, 32)
  dim3 g3072(FF_ / 128, MM / 128);  // (24, 32)

  gemm_bf16_kernel<0><<<g768, 256, 0, stream>>>(h_bf, wq_bf, bq, nullptr, nullptr, q_bf, MM, DD, DD);
  gemm_bf16_kernel<0><<<g768, 256, 0, stream>>>(h_bf, wk_bf, bk, nullptr, nullptr, k_bf, MM, DD, DD);
  gemm_bf16_kernel<0><<<g768, 256, 0, stream>>>(h_bf, wv_bf, bv, nullptr, nullptr, v_bf, MM, DD, DD);

  attn_kernel<<<dim3(TT / 64, BB * HH), 128, 0, stream>>>(q_bf, k_bf, v_bf, o_bf);

  gemm_bf16_kernel<2><<<g768, 256, 0, stream>>>(o_bf, wo_bf, bo, x, x1, nullptr, MM, DD, DD);

  ln_kernel<<<MM, 256, 0, stream>>>(x1, ln2_g, ln2_b, h_bf);

  gemm_bf16_kernel<1><<<g3072, 256, 0, stream>>>(h_bf, w1_bf, b1, nullptr, nullptr, m1_bf, MM, FF_, DD);
  gemm_bf16_kernel<2><<<g768, 256, 0, stream>>>(m1_bf, w2_bf, b2, x1, out, nullptr, MM, DD, FF_);
}
